// SelfAttention_26431228740295
// MI455X (gfx1250) — matrix operand order repaired and checked by reading, not compiled, not run on hardware
//
#include <hip/hip_runtime.h>

typedef __attribute__((ext_vector_type(16))) __bf16 v16bf;
typedef __attribute__((ext_vector_type(8)))  __bf16 v8bf;
typedef __attribute__((ext_vector_type(8)))  float  v8f;

// Matches the builtin's parameter type from the clang diagnostic:
// '__attribute__((__vector_size__(4 * sizeof(int)))) int __device__ *'
typedef int v4i __attribute__((vector_size(16)));
typedef __attribute__((address_space(1))) v4i* GV4P;   // global
typedef __attribute__((address_space(3))) v4i* LV4P;   // LDS

#define BATCH   8
#define SEQ     1024
#define EMBED   768
#define HEADS   12
#define HDIM    64
#define M_ROWS  (BATCH * SEQ)      // 8192
#define QKV_N   (3 * EMBED)        // 2304
#define SCALE_F 0.125f             // 64^-0.5

// ---------------- WMMA fragment loaders (bf16, 16x16x32) ----------------
// A: 16x32 (MxK). lane L: row M = L&15; element e -> K = k0+16*half+e
// (16 contiguous).
__device__ __forceinline__ v16bf load_a_frag(const __bf16* A, int lda, int m0,
                                             int k0, int lane) {
    int half = (lane >> 4) & 1;
    int row  = lane & 15;
    const __bf16* p = A + (size_t)(m0 + row) * lda + k0 + 16 * half;
    v8bf lo = *(const v8bf*)(p);
    v8bf hi = *(const v8bf*)(p + 8);
    v16bf a;
#pragma unroll
    for (int i = 0; i < 8; ++i) { a[i] = lo[i]; a[8 + i] = hi[i]; }
    return a;
}

// B: 32x16 (KxN), sourced from Bt stored row-major [n][k] (i.e. B transposed).
// lane L: col N = L&15; element e -> K = k0 + 16*half + e  (16 contiguous).
__device__ __forceinline__ v16bf load_b_frag(const __bf16* Bt, int ldb, int n0,
                                             int k0, int lane) {
    int half = (lane >> 4) & 1;
    int col  = lane & 15;
    const __bf16* p = Bt + (size_t)(n0 + col) * ldb + k0 + 16 * half;
    return *(const v16bf*)(p);
}

__device__ __forceinline__ v8f wmma_bf16(v16bf a, v16bf b, v8f c) {
    return __builtin_amdgcn_wmma_f32_16x16x32_bf16(false, a, false, b,
                                                   (short)0, c, false, false);
}

// Stage a contiguous strip of bf16 from global into LDS, block-cooperative.
// Uses the gfx1250 async-to-LDS path (ASYNCcnt) when available, else direct.
__device__ __forceinline__ void stage_to_lds(const __bf16* src, __bf16* lds,
                                             int n_elems, int tid,
                                             int nthreads) {
    int chunks = n_elems / 8;  // 16B chunks
    for (int c = tid; c < chunks; c += nthreads) {
#if __has_builtin(__builtin_amdgcn_global_load_async_to_lds_b128)
        __builtin_amdgcn_global_load_async_to_lds_b128(
            (GV4P)(src + c * 8), (LV4P)(lds + c * 8), 0, 0);
#else
        *(v8bf*)(lds + c * 8) = *(const v8bf*)(src + c * 8);
#endif
    }
#if __has_builtin(__builtin_amdgcn_global_load_async_to_lds_b128) && \
    __has_builtin(__builtin_amdgcn_s_wait_asynccnt)
    __builtin_amdgcn_s_wait_asynccnt(0);
#endif
}

// ---------------- conversion / transpose kernels ----------------
__global__ void cvt_f32_bf16_kernel(const float* __restrict__ src,
                                    __bf16* __restrict__ dst, int n) {
    for (int i = blockIdx.x * blockDim.x + threadIdx.x; i < n;
         i += gridDim.x * blockDim.x)
        dst[i] = (__bf16)src[i];
}

// w: [K][N] f32 -> wt: [N][K] bf16
__global__ void transpose_cvt_kernel(const float* __restrict__ w,
                                     __bf16* __restrict__ wt, int K, int N) {
    int total = K * N;
    for (int i = blockIdx.x * blockDim.x + threadIdx.x; i < total;
         i += gridDim.x * blockDim.x) {
        int n = i / K, k = i % K;
        wt[i] = (__bf16)w[(size_t)k * N + n];
    }
}

// ---------------- QKV projection GEMM ----------------
// 32x32 tile per wave; 8 waves/block share one 32-row A strip staged in LDS.
__device__ __forceinline__ void qkv_store(v8f acc, int m_base, int colg,
                                          const float* __restrict__ bias,
                                          int half, __bf16* __restrict__ Qb,
                                          __bf16* __restrict__ Kb,
                                          __bf16* __restrict__ Vtb) {
    float bv = bias[colg];
    int s = colg / EMBED;               // 0=Q 1=K 2=V
    int rem = colg % EMBED;
    int h = rem >> 6, d = rem & 63;
#pragma unroll
    for (int r = 0; r < 8; ++r) {
        int m  = m_base + r + 8 * half;
        int b_ = m >> 10, n = m & 1023;
        size_t bh = (size_t)b_ * HEADS + h;
        __bf16 o = (__bf16)(acc[r] + bv);
        if (s == 0)      Qb [(bh * SEQ + n) * HDIM + d] = o;
        else if (s == 1) Kb [(bh * SEQ + n) * HDIM + d] = o;
        else             Vtb[(bh * HDIM + d) * SEQ + n] = o;
    }
}

__global__ void qkv_gemm_kernel(const __bf16* __restrict__ xb,
                                const __bf16* __restrict__ wqT,
                                const float* __restrict__ bias,
                                __bf16* __restrict__ Qb,
                                __bf16* __restrict__ Kb,
                                __bf16* __restrict__ Vtb) {
    __shared__ __bf16 Atile[32 * EMBED];           // 48 KB
    int tid  = threadIdx.x;
    int lane = tid & 31;
    int wave = tid >> 5;
    int gid  = blockIdx.x * 8 + wave;
    int nt   = gid % (QKV_N / 32);                 // 72 (div by 8 -> shared mt)
    int mt   = gid / (QKV_N / 32);                 // 256
    int m0 = mt * 32, c0 = nt * 32;

    stage_to_lds(xb + (size_t)m0 * EMBED, Atile, 32 * EMBED, tid, 256);
    __syncthreads();

    v8f acc00 = {}, acc01 = {}, acc10 = {}, acc11 = {};
#pragma unroll 4
    for (int k0 = 0; k0 < EMBED; k0 += 32) {
        v16bf a0 = load_a_frag(Atile, EMBED, 0,  k0, lane);
        v16bf a1 = load_a_frag(Atile, EMBED, 16, k0, lane);
        v16bf b0 = load_b_frag(wqT, EMBED, c0,      k0, lane);
        v16bf b1 = load_b_frag(wqT, EMBED, c0 + 16, k0, lane);
        acc00 = wmma_bf16(a0, b0, acc00);
        acc01 = wmma_bf16(a0, b1, acc01);
        acc10 = wmma_bf16(a1, b0, acc10);
        acc11 = wmma_bf16(a1, b1, acc11);
    }

    int half = lane >> 4;
    int cl = lane & 15;
    qkv_store(acc00, m0,      c0 + cl,      bias, half, Qb, Kb, Vtb);
    qkv_store(acc01, m0,      c0 + 16 + cl, bias, half, Qb, Kb, Vtb);
    qkv_store(acc10, m0 + 16, c0 + cl,      bias, half, Qb, Kb, Vtb);
    qkv_store(acc11, m0 + 16, c0 + 16 + cl, bias, half, Qb, Kb, Vtb);
}

// ---------------- attention: scores + softmax + P@V ----------------
// One block (128 thr = 4 waves) per (b,h,qtile): 16 queries x 1024 keys.
__global__ void attn_kernel(const __bf16* __restrict__ Qb,
                            const __bf16* __restrict__ Kb,
                            const __bf16* __restrict__ Vtb,
                            float* __restrict__ attn,
                            __bf16* __restrict__ Ob) {
    extern __shared__ float smem[];
    float* S    = smem;                 // 16*1024
    float* red  = smem + 16 * 1024;     // 16*8
    float* rowv = red + 16 * 8;         // 16

    int tid  = threadIdx.x;
    int lane = tid & 31;
    int wave = tid >> 5;
    size_t bh = blockIdx.x >> 6;        // 0..95
    int qt = blockIdx.x & 63;
    int m0 = qt * 16;

    const __bf16* Qbase = Qb  + bh * SEQ * HDIM;
    const __bf16* Kbase = Kb  + bh * SEQ * HDIM;
    const __bf16* Vbase = Vtb + bh * HDIM * SEQ;

    // Phase 1: S = scale * Q Kt. Q fragments are kt-invariant -> hoisted.
    {
        v16bf qa0 = load_a_frag(Qbase, HDIM, m0, 0,  lane);
        v16bf qa1 = load_a_frag(Qbase, HDIM, m0, 32, lane);
        int half = lane >> 4;
        for (int kt = wave; kt < SEQ / 16; kt += 4) {
            v8f acc = {};
            v16bf b0 = load_b_frag(Kbase, HDIM, kt * 16, 0,  lane);
            v16bf b1 = load_b_frag(Kbase, HDIM, kt * 16, 32, lane);
            acc = wmma_bf16(qa0, b0, acc);
            acc = wmma_bf16(qa1, b1, acc);
            int colL = kt * 16 + (lane & 15);
#pragma unroll
            for (int r = 0; r < 8; ++r)
                S[(r + 8 * half) * SEQ + colL] = acc[r] * SCALE_F;
        }
    }
    __syncthreads();

    // Phase 2: softmax over 1024 per row; 8 chunks of 128 per row.
    {
        int r  = tid & 15;
        int ch = tid >> 4;
        float* Sr = S + r * SEQ + ch * 128;
        float mx = -3.4e38f;
        for (int i = 0; i < 128; ++i) mx = fmaxf(mx, Sr[i]);
        red[r * 8 + ch] = mx;
        __syncthreads();
        if (ch == 0) {
            float m = red[r * 8];
            for (int j = 1; j < 8; ++j) m = fmaxf(m, red[r * 8 + j]);
            rowv[r] = m;
        }
        __syncthreads();
        float rm = rowv[r];
        float sum = 0.f;
        for (int i = 0; i < 128; ++i) {
            float e = __expf(Sr[i] - rm);
            Sr[i] = e;
            sum += e;
        }
        red[r * 8 + ch] = sum;
        __syncthreads();
        if (ch == 0) {
            float s = 0.f;
            for (int j = 0; j < 8; ++j) s += red[r * 8 + j];
            rowv[r] = s;
        }
        __syncthreads();
        float inv = 1.0f / rowv[r];
        float* aout = attn + (bh * SEQ + (m0 + r)) * SEQ + ch * 128;
        for (int i = 0; i < 128; ++i) {
            float p = Sr[i] * inv;
            Sr[i] = p;
            aout[i] = p;
        }
    }
    __syncthreads();

    // Phase 3: O = P @ V ; wave w owns d-tile [w*16, w*16+16).
    {
        int dt = wave;                   // 0..3
        int half = lane >> 4;
        int col = lane & 15;
        const __bf16* Vp = Vbase + (size_t)(dt * 16 + col) * SEQ + 16 * half;
        int mrow = lane & 15;
        v8f acc = {};
        for (int kt = 0; kt < SEQ; kt += 32) {
            v16bf a;
            const float* Sp = S + mrow * SEQ + kt + 16 * half;
#pragma unroll
            for (int e = 0; e < 8; ++e) {
                a[e]     = (__bf16)Sp[e];
                a[8 + e] = (__bf16)Sp[8 + e];
            }
            v16bf b = *(const v16bf*)(Vp + kt);
            acc = wmma_bf16(a, b, acc);
        }
        int b_ = (int)(bh / HEADS), h = (int)(bh % HEADS);
#pragma unroll
        for (int r = 0; r < 8; ++r) {
            int q = m0 + r + 8 * half;
            Ob[((size_t)b_ * SEQ + q) * EMBED + h * HDIM + dt * 16 + col] =
                (__bf16)acc[r];
        }
    }
}

// ---------------- output projection GEMM ----------------
// 32x32 tile per wave; 8 waves/block share one 32-row A strip staged in LDS.
__global__ void proj_gemm_kernel(const __bf16* __restrict__ Ob,
                                 const __bf16* __restrict__ wpT,
                                 const float* __restrict__ bias,
                                 float* __restrict__ out) {
    __shared__ __bf16 Atile[32 * EMBED];           // 48 KB
    int tid  = threadIdx.x;
    int lane = tid & 31;
    int wave = tid >> 5;
    int gid  = blockIdx.x * 8 + wave;
    int nt   = gid % (EMBED / 32);                 // 24 (div by 8 -> shared mt)
    int mt   = gid / (EMBED / 32);                 // 256
    int m0 = mt * 32, n0 = nt * 32;

    stage_to_lds(Ob + (size_t)m0 * EMBED, Atile, 32 * EMBED, tid, 256);
    __syncthreads();

    v8f acc00 = {}, acc01 = {}, acc10 = {}, acc11 = {};
#pragma unroll 4
    for (int k0 = 0; k0 < EMBED; k0 += 32) {
        v16bf a0 = load_a_frag(Atile, EMBED, 0,  k0, lane);
        v16bf a1 = load_a_frag(Atile, EMBED, 16, k0, lane);
        v16bf b0 = load_b_frag(wpT, EMBED, n0,      k0, lane);
        v16bf b1 = load_b_frag(wpT, EMBED, n0 + 16, k0, lane);
        acc00 = wmma_bf16(a0, b0, acc00);
        acc01 = wmma_bf16(a0, b1, acc01);
        acc10 = wmma_bf16(a1, b0, acc10);
        acc11 = wmma_bf16(a1, b1, acc11);
    }

    int half = lane >> 4;
    int cl = lane & 15;
#pragma unroll
    for (int r = 0; r < 8; ++r) {
        int mA = m0 + r + 8 * half;
        int mB = m0 + 16 + r + 8 * half;
        out[(size_t)mA * EMBED + n0 + cl]      = acc00[r] + bias[n0 + cl];
        out[(size_t)mA * EMBED + n0 + 16 + cl] = acc01[r] + bias[n0 + 16 + cl];
        out[(size_t)mB * EMBED + n0 + cl]      = acc10[r] + bias[n0 + cl];
        out[(size_t)mB * EMBED + n0 + 16 + cl] = acc11[r] + bias[n0 + 16 + cl];
    }
}

// ---------------- launch ----------------
extern "C" void kernel_launch(void* const* d_in, const int* in_sizes, int n_in,
                              void* d_out, int out_size, void* d_ws,
                              size_t ws_size, hipStream_t stream) {
    (void)in_sizes; (void)n_in; (void)out_size; (void)ws_size;
    const float* x      = (const float*)d_in[0];
    const float* qkv_w  = (const float*)d_in[1];
    const float* qkv_b  = (const float*)d_in[2];
    const float* proj_w = (const float*)d_in[3];
    const float* proj_b = (const float*)d_in[4];

    float* out  = (float*)d_out;                       // [8,1024,768]
    float* attn = out + (size_t)BATCH * SEQ * EMBED;   // [8,12,1024,1024]

    char* ws = (char*)d_ws;
    const size_t XB   = (size_t)M_ROWS * EMBED * 2;    // 12,582,912
    const size_t WQT  = (size_t)QKV_N * EMBED * 2;     //  3,538,944
    const size_t WPT  = (size_t)EMBED * EMBED * 2;     //  1,179,648
    __bf16* xb  = (__bf16*)(ws);
    __bf16* wqT = (__bf16*)(ws + XB);
    __bf16* wpT = (__bf16*)(ws + XB + WQT);
    __bf16* Qb  = (__bf16*)(ws + XB + WQT + WPT);
    __bf16* Kb  = (__bf16*)(ws + XB + WQT + WPT + XB);
    __bf16* Vtb = (__bf16*)(ws + XB + WQT + WPT + 2 * XB);
    __bf16* Ob  = (__bf16*)(ws + XB + WQT + WPT + 3 * XB);

    cvt_f32_bf16_kernel<<<2048, 256, 0, stream>>>(x, xb, M_ROWS * EMBED);
    transpose_cvt_kernel<<<(EMBED * QKV_N + 255) / 256, 256, 0, stream>>>(
        qkv_w, wqT, EMBED, QKV_N);
    transpose_cvt_kernel<<<(EMBED * EMBED + 255) / 256, 256, 0, stream>>>(
        proj_w, wpT, EMBED, EMBED);

    // 256*72 wave-tiles (32x32 each) / 8 waves per block
    qkv_gemm_kernel<<<(M_ROWS / 32) * (QKV_N / 32) / 8, 256, 0, stream>>>(
        xb, wqT, qkv_b, Qb, Kb, Vtb);

    // one block per (b,h,qtile); 66112 bytes dynamic LDS
    attn_kernel<<<BATCH * HEADS * (SEQ / 16), 128,
                  (16 * 1024 + 16 * 8 + 16) * sizeof(float), stream>>>(
        Qb, Kb, Vtb, attn, Ob);

    // 256*24 wave-tiles (32x32 each) / 8 waves per block
    proj_gemm_kernel<<<(M_ROWS / 32) * (EMBED / 32) / 8, 256, 0, stream>>>(
        Ob, wpT, proj_b, out);
}
